// Head_42030549958907
// MI455X (gfx1250) — compile-verified, matrix operand order checked by reading, not run on hardware
//
#include <hip/hip_runtime.h>
#include <hip/hip_bf16.h>

// ---------------------------------------------------------------------------
// Single-head causal attention for MI455X (gfx1250, wave32, WMMA).
//   B=4, T=2048, D=1024, H=64.
// Path: fp32 -> bf16 operands, v_wmma_f32_16x16x32_bf16 (fp32 accum),
// flash-style online softmax in fp32, fp32 output.
// ---------------------------------------------------------------------------

typedef __attribute__((ext_vector_type(16))) __bf16 v16bf;
typedef __attribute__((ext_vector_type(8)))  __bf16 v8bf;
typedef __attribute__((ext_vector_type(8)))  float  v8f;

#define WMMA_BF16(A, B, C) \
    __builtin_amdgcn_wmma_f32_16x16x32_bf16(false, (A), false, (B), (short)0, (C), false, false)

// Constants of this problem
constexpr int Tt = 2048;   // sequence length
constexpr int Dd = 1024;   // model dim
constexpr int Hh = 64;     // head dim
constexpr int Bb = 4;      // batch

// ---------------------------------------------------------------------------
// Tile loaders for the CDNA5 16-bit WMMA operand layout (wave32):
//   lane 0..15 :  index = lane,    elements 0..7 -> K 0..7,  8..15 -> K 16..23
//   lane 16..31:  index = lane-16, elements 0..7 -> K 8..15, 8..15 -> K 24..31
// A-tiles use index=M (row), B-tiles use index=N (col of output); with B
// stored "transposed" (rows indexed by N, contiguous in K) both loads share
// this addressing, so K-matrix rows and Vt rows load exactly like A-tiles.
// ---------------------------------------------------------------------------
__device__ inline v16bf load_tile_bf16(const __bf16* base, int row0, int k0, int ld) {
    const int lane = threadIdx.x & 31;
    const int r    = lane & 15;
    const int kb   = (lane >> 4) << 3;              // 0 or 8
    const __bf16* p = base + (size_t)(row0 + r) * ld + k0 + kb;
    v8bf lo = *(const v8bf*)(p);                    // K kb..kb+7
    v8bf hi = *(const v8bf*)(p + 16);               // K kb+16..kb+23
    return __builtin_shufflevector(lo, hi, 0,1,2,3,4,5,6,7,8,9,10,11,12,13,14,15);
}

__device__ inline v16bf load_tile_f32_as_bf16(const float* base, int row0, int k0, int ld) {
    const int lane = threadIdx.x & 31;
    const int r    = lane & 15;
    const int kb   = (lane >> 4) << 3;
    const float* p = base + (size_t)(row0 + r) * ld + k0 + kb;
    v8f lo = *(const v8f*)(p);
    v8f hi = *(const v8f*)(p + 16);
    v16bf out;
#pragma unroll
    for (int i = 0; i < 8; ++i) { out[i] = (__bf16)lo[i]; out[i + 8] = (__bf16)hi[i]; }
    return out;
}

// Reductions across the 16-lane half that owns one C/D row (masks < 16 never
// cross the half boundary in wave32).
__device__ inline float half16_max(float v) {
#pragma unroll
    for (int m = 1; m <= 8; m <<= 1) v = fmaxf(v, __shfl_xor(v, m, 32));
    return v;
}
__device__ inline float half16_sum(float v) {
#pragma unroll
    for (int m = 1; m <= 8; m <<= 1) v += __shfl_xor(v, m, 32);
    return v;
}

// ---------------------------------------------------------------------------
// Kernel 0: Wt[w][h][d] = (bf16) W_w[d][h]   (w in {q,k,v})
// Makes projection B-tiles contiguous in the K (=d) dimension.
// ---------------------------------------------------------------------------
__global__ __launch_bounds__(256) void wt_transpose_kernel(
    const float* __restrict__ Wq, const float* __restrict__ Wk,
    const float* __restrict__ Wv, __bf16* __restrict__ Wt) {
    int i = blockIdx.x * 256 + threadIdx.x;          // 3*64*1024 total
    int d = i & (Dd - 1);
    int h = (i >> 10) & (Hh - 1);
    int w = i >> 16;
    const float* W = (w == 0) ? Wq : (w == 1) ? Wk : Wv;
    Wt[i] = (__bf16)W[(size_t)d * Hh + h];
}

// ---------------------------------------------------------------------------
// Kernel 1: fused QKV projection.  One wave per 16-token tile.
// acc[w*4+nt] : 16x16 fp32 accumulators, 3 matrices x 4 N-tiles (H=64).
// Writes Q,K as bf16 [token][h]; V transposed as bf16 Vt[b][h][t].
// ---------------------------------------------------------------------------
__global__ __launch_bounds__(256) void qkv_proj_kernel(
    const float* __restrict__ x, const __bf16* __restrict__ Wt,
    __bf16* __restrict__ Q, __bf16* __restrict__ K, __bf16* __restrict__ Vt) {
    const int wave = threadIdx.x >> 5;
    const int lane = threadIdx.x & 31;
    const int m0   = (blockIdx.x * 8 + wave) * 16;   // flattened B*T row tile

    v8f acc[12];
#pragma unroll
    for (int i = 0; i < 12; ++i) acc[i] = (v8f){0.f,0.f,0.f,0.f,0.f,0.f,0.f,0.f};

    for (int kk = 0; kk < Dd; kk += 32) {
        v16bf a = load_tile_f32_as_bf16(x, m0, kk, Dd);
#pragma unroll
        for (int w = 0; w < 3; ++w) {
            const __bf16* wt = Wt + (size_t)w * Hh * Dd;
#pragma unroll
            for (int nt = 0; nt < 4; ++nt) {
                v16bf b = load_tile_bf16(wt, nt * 16, kk, Dd);
                acc[w * 4 + nt] = WMMA_BF16(a, b, acc[w * 4 + nt]);
            }
        }
    }

    // C/D layout: VGPR r, lanes 0-15 -> row r, lanes 16-31 -> row r+8; col = lane&15.
    const int colb = lane & 15;
    const int rofs = (lane < 16) ? 0 : 8;
    const int b    = m0 >> 11;                       // batch (T=2048)
#pragma unroll
    for (int r = 0; r < 8; ++r) {
        const int row = m0 + r + rofs;               // flattened token index
        const int t   = row & (Tt - 1);
#pragma unroll
        for (int nt = 0; nt < 4; ++nt) {
            const int h = nt * 16 + colb;
            Q[(size_t)row * Hh + h] = (__bf16)acc[0 + nt][r];
            K[(size_t)row * Hh + h] = (__bf16)acc[4 + nt][r];
            Vt[((size_t)b * Hh + h) * Tt + t] = (__bf16)acc[8 + nt][r];
        }
    }
}

// ---------------------------------------------------------------------------
// Kernel 2: flash-style causal attention.  One wave per 16-row Q tile.
// Tiles interleaved (tile = wave*64 + blockIdx.x) so each workgroup carries
// one Q tile from each half of every batch.
// ---------------------------------------------------------------------------
__global__ __launch_bounds__(256) void attn_kernel(
    const __bf16* __restrict__ Q, const __bf16* __restrict__ K,
    const __bf16* __restrict__ Vt, float* __restrict__ out) {
    __shared__ __align__(16) __bf16 lds_p[8][16 * 32];   // per-wave P staging (1 KB)

    const int wave = threadIdx.x >> 5;
    const int lane = threadIdx.x & 31;
    const int tile = wave * 64 + blockIdx.x;             // 512 tiles total
    const int b    = tile >> 7;                          // 128 tiles per batch
    const int q0   = (tile & 127) << 4;

    const __bf16* Qb  = Q  + (size_t)b * Tt * Hh;
    const __bf16* Kb  = K  + (size_t)b * Tt * Hh;
    const __bf16* Vtb = Vt + (size_t)b * Hh * Tt;

    // Q tile in A-layout, two 32-wide K-steps over H=64.
    const v16bf qa0 = load_tile_bf16(Qb, q0, 0,  Hh);
    const v16bf qa1 = load_tile_bf16(Qb, q0, 32, Hh);

    v8f acc0 = {}, acc1 = {}, acc2 = {}, acc3 = {};
    v8f m_i, l_i;
#pragma unroll
    for (int r = 0; r < 8; ++r) { m_i[r] = -INFINITY; l_i[r] = 0.f; }

    const float scale  = 0.125f;                         // H^-1/2
    const int   rbase  = q0 + ((lane < 16) ? 0 : 8);     // row of stat element r = rbase + r
    const int   colsub = lane & 15;
    __bf16* myp = &lds_p[wave][0];

    for (int s0 = 0; s0 < q0 + 16; s0 += 32) {
        // ---- S = Q K^T for two 16-col subtiles -------------------------------
        v8f s_sub[2];
#pragma unroll
        for (int sub = 0; sub < 2; ++sub) {
            v16bf kb0 = load_tile_bf16(Kb, s0 + sub * 16, 0,  Hh);
            v16bf kb1 = load_tile_bf16(Kb, s0 + sub * 16, 32, Hh);
            v8f s = {};
            s = WMMA_BF16(qa0, kb0, s);
            s = WMMA_BF16(qa1, kb1, s);
            s_sub[sub] = s;
        }
        // ---- scale + causal mask + online softmax ----------------------------
        const int col0 = s0 + colsub;
        v8f rm;
#pragma unroll
        for (int r = 0; r < 8; ++r) {
            const int row = rbase + r;
            float x0 = (col0      <= row) ? s_sub[0][r] * scale : -INFINITY;
            float x1 = (col0 + 16 <= row) ? s_sub[1][r] * scale : -INFINITY;
            s_sub[0][r] = x0; s_sub[1][r] = x1;
            rm[r] = fmaxf(x0, x1);
        }
#pragma unroll
        for (int r = 0; r < 8; ++r) rm[r] = half16_max(rm[r]);
        v8f corr, rs;
#pragma unroll
        for (int r = 0; r < 8; ++r) {
            const float mn = fmaxf(m_i[r], rm[r]);       // finite: diagonal always live
            corr[r] = __expf(m_i[r] - mn);
            m_i[r]  = mn;
            const float p0 = __expf(s_sub[0][r] - mn);
            const float p1 = __expf(s_sub[1][r] - mn);
            s_sub[0][r] = p0; s_sub[1][r] = p1;
            rs[r] = p0 + p1;
        }
#pragma unroll
        for (int r = 0; r < 8; ++r) rs[r] = half16_sum(rs[r]);
#pragma unroll
        for (int r = 0; r < 8; ++r) {
            l_i[r] = l_i[r] * corr[r] + rs[r];
            acc0[r] *= corr[r]; acc1[r] *= corr[r];
            acc2[r] *= corr[r]; acc3[r] *= corr[r];
        }
        // ---- P: C/D layout -> LDS (row-major 16x32) -> A-layout -------------
#pragma unroll
        for (int sub = 0; sub < 2; ++sub)
#pragma unroll
            for (int r = 0; r < 8; ++r) {
                const int row = r + ((lane < 16) ? 0 : 8);
                myp[row * 32 + sub * 16 + colsub] = (__bf16)s_sub[sub][r];
            }
        v16bf pa;
        {
            const int rr = lane & 15;
            const int kb = (lane >> 4) << 3;
            const __bf16* p = myp + rr * 32 + kb;
            v8bf lo = *(const v8bf*)(p);
            v8bf hi = *(const v8bf*)(p + 16);
            pa = __builtin_shufflevector(lo, hi, 0,1,2,3,4,5,6,7,8,9,10,11,12,13,14,15);
        }
        // ---- O += P * V  (B-tiles from Vt, contiguous in s) ------------------
        acc0 = WMMA_BF16(pa, load_tile_bf16(Vtb, 0,  s0, Tt), acc0);
        acc1 = WMMA_BF16(pa, load_tile_bf16(Vtb, 16, s0, Tt), acc1);
        acc2 = WMMA_BF16(pa, load_tile_bf16(Vtb, 32, s0, Tt), acc2);
        acc3 = WMMA_BF16(pa, load_tile_bf16(Vtb, 48, s0, Tt), acc3);
    }

    // ---- normalize + store fp32 ---------------------------------------------
    float* ob = out + (size_t)b * Tt * Hh;
#pragma unroll
    for (int r = 0; r < 8; ++r) {
        const int row = q0 + r + ((lane < 16) ? 0 : 8);
        const float inv = 1.0f / l_i[r];
        float* orow = ob + (size_t)row * Hh + colsub;
        orow[0]  = acc0[r] * inv;
        orow[16] = acc1[r] * inv;
        orow[32] = acc2[r] * inv;
        orow[48] = acc3[r] * inv;
    }
}

// ---------------------------------------------------------------------------
// Host-side launcher
// ---------------------------------------------------------------------------
extern "C" void kernel_launch(void* const* d_in, const int* in_sizes, int n_in,
                              void* d_out, int out_size, void* d_ws, size_t ws_size,
                              hipStream_t stream) {
    (void)in_sizes; (void)n_in; (void)out_size; (void)ws_size;
    const float* x  = (const float*)d_in[0];
    const float* Wq = (const float*)d_in[1];
    const float* Wk = (const float*)d_in[2];
    const float* Wv = (const float*)d_in[3];
    float* out = (float*)d_out;

    char* ws = (char*)d_ws;
    const size_t WT_BYTES = (size_t)3 * Hh * Dd * sizeof(__bf16);     // 384 KB
    const size_t QK_BYTES = (size_t)Bb * Tt * Hh * sizeof(__bf16);    // 1 MB each
    __bf16* Wt  = (__bf16*)(ws);
    __bf16* Qb  = (__bf16*)(ws + WT_BYTES);
    __bf16* Kb  = (__bf16*)(ws + WT_BYTES + QK_BYTES);
    __bf16* Vtb = (__bf16*)(ws + WT_BYTES + 2 * QK_BYTES);

    wt_transpose_kernel<<<(3 * Hh * Dd) / 256, 256, 0, stream>>>(Wq, Wk, Wv, Wt);
    qkv_proj_kernel<<<64, 256, 0, stream>>>(x, Wt, Qb, Kb, Vtb);
    attn_kernel<<<64, 256, 0, stream>>>(Qb, Kb, Vtb, out);
}
